// GlmMoE_24756191494627
// MI455X (gfx1250) — matrix operand order repaired and checked by reading, not compiled, not run on hardware
//
#include <hip/hip_runtime.h>

// ---------------- problem constants ----------------
#define T_TOK 2048   // B*S
#define HDIM  1024
#define IDIM  1024
#define SDIM  1024
#define NEXP  16
#define TOPK  4
#define BK    32     // K-step per stage (one bf16 WMMA deep)
#define BM    128    // block tile M
#define BN    64     // block tile N
#define LDP   40     // LDS row pitch in ushorts (bank-stagger + 16B align)
#define MTILES (T_TOK / BM)   // 16 M-tiles over all tokens

typedef __bf16        v16bf __attribute__((ext_vector_type(16)));
typedef float         v8f   __attribute__((ext_vector_type(8)));
typedef float         vf4   __attribute__((ext_vector_type(4)));
typedef unsigned int  vu4   __attribute__((ext_vector_type(4)));

union FragAB { v16bf v; vu4 q[2]; };

// ---------------- helpers ----------------
__device__ __forceinline__ unsigned short bf1(float f) {
    unsigned int u = __float_as_uint(f);
    return (unsigned short)((u + 0x7FFFu + ((u >> 16) & 1u)) >> 16);
}
__device__ __forceinline__ unsigned int bfpair(float a, float b) {
    unsigned int ua = __float_as_uint(a);
    ua = ua + 0x7FFFu + ((ua >> 16) & 1u);
    unsigned int ub = __float_as_uint(b);
    ub = ub + 0x7FFFu + ((ub >> 16) & 1u);
    return (ua >> 16) | (ub & 0xFFFF0000u);
}
__device__ __forceinline__ vu4 pack8(vf4 a, vf4 b) {
    vu4 r;
    r.x = bfpair(a.x, a.y);
    r.y = bfpair(a.z, a.w);
    r.z = bfpair(b.x, b.y);
    r.w = bfpair(b.z, b.w);
    return r;
}
// fast silu: one v_rcp_f32 instead of a full IEEE divide chain
__device__ __forceinline__ float siluf(float v) {
    return v * __builtin_amdgcn_rcpf(1.f + __expf(-v));
}
__device__ __forceinline__ v8f z8() { v8f z = {0,0,0,0,0,0,0,0}; return z; }

// ----- register staging (global bf16 -> regs -> LDS bf16, pure copies) -----
struct ARB { vu4 q[2]; };   // 16 bf16 (one A row chunk)
struct BRB { vu4 q; };      // 8 bf16 (one B row chunk)

__device__ __forceinline__ void ld_abf(ARB& r, const unsigned short* p) {
    r.q[0] = *(const vu4*)p;
    r.q[1] = *(const vu4*)(p + 8);
}
__device__ __forceinline__ void st_abf(unsigned short* As, int row, int col, const ARB& r) {
    *(vu4*)(As + row * LDP + col)     = r.q[0];
    *(vu4*)(As + row * LDP + col + 8) = r.q[1];
}
__device__ __forceinline__ void ld_bbf(BRB& r, const unsigned short* p) {
    r.q = *(const vu4*)p;
}
__device__ __forceinline__ void st_bbf(unsigned short* Bs, int row, int col, const BRB& r) {
    *(vu4*)(Bs + row * LDP + col) = r.q;
}

// ----- one K=32 step: wave (wmq,wnq) computes a 32x32 patch: 4 WMMAs -----
__device__ __forceinline__ void mma4(const unsigned short* As, const unsigned short* Bs,
                                     int wmq, int wnq, int lane, v8f (&acc)[2][2]) {
    const int half = lane >> 4, lm = lane & 15;
    FragAB a[2], b[2];
#pragma unroll
    for (int mt = 0; mt < 2; ++mt) {
        const unsigned short* ap = As + (wmq * 32 + mt * 16 + lm) * LDP + half * 8;
        a[mt].q[0] = *(const vu4*)ap;          // K = kbase..kbase+7
        a[mt].q[1] = *(const vu4*)(ap + 16);   // K = kbase+16..kbase+23
    }
#pragma unroll
    for (int nt = 0; nt < 2; ++nt) {
        const unsigned short* bp = Bs + (wnq * 32 + nt * 16 + lm) * LDP + half * 8;
        b[nt].q[0] = *(const vu4*)bp;
        b[nt].q[1] = *(const vu4*)(bp + 16);
    }
#pragma unroll
    for (int mt = 0; mt < 2; ++mt)
#pragma unroll
        for (int nt = 0; nt < 2; ++nt)
            acc[mt][nt] = __builtin_amdgcn_wmma_f32_16x16x32_bf16(
                false, a[mt].v, false, b[nt].v, (short)0, acc[mt][nt], false, false);
}

// ---------------- bulk f32 -> bf16 preconversion (streaming) ----------------
__global__ __launch_bounds__(256) void k_cvt(const float* __restrict__ src,
                                             unsigned short* __restrict__ dst, int n) {
    int i = (blockIdx.x * 256 + threadIdx.x) * 8;
    if (i < n) {
        vf4 a = *(const vf4*)(src + i);
        vf4 b = *(const vf4*)(src + i + 4);
        *(vu4*)(dst + i) = pack8(a, b);
    }
}

// ---------------- routing kernels ----------------
__global__ __launch_bounds__(64) void k_init(int* counts, int* cursor) {
    int t = threadIdx.x;
    if (t < NEXP) counts[t] = 0;
    else if (t < 2 * NEXP) cursor[t - NEXP] = 0;
}

// one wave per token: logits -> softmax -> top4 -> renorm
__global__ __launch_bounds__(256) void k_router(const float* __restrict__ x,
                                                const float* __restrict__ gw,
                                                int* __restrict__ topi,
                                                float* __restrict__ topv,
                                                int* __restrict__ counts) {
    const int lane = threadIdx.x & 31;
    const int t = blockIdx.x * 8 + (threadIdx.x >> 5);
    float acc[NEXP];
#pragma unroll
    for (int e = 0; e < NEXP; ++e) acc[e] = 0.f;
    for (int h = lane; h < HDIM; h += 32) {
        float xv = x[(size_t)t * HDIM + h];
#pragma unroll
        for (int e = 0; e < NEXP; ++e)
            acc[e] = fmaf(xv, gw[(size_t)e * HDIM + h], acc[e]);
    }
#pragma unroll
    for (int off = 16; off >= 1; off >>= 1) {
#pragma unroll
        for (int e = 0; e < NEXP; ++e) acc[e] += __shfl_xor(acc[e], off, 32);
    }
    if (lane == 0) {
        float m = acc[0];
#pragma unroll
        for (int e = 1; e < NEXP; ++e) m = fmaxf(m, acc[e]);
        float p[NEXP], s = 0.f;
#pragma unroll
        for (int e = 0; e < NEXP; ++e) { p[e] = __expf(acc[e] - m); s += p[e]; }
        float rs = __builtin_amdgcn_rcpf(s);
#pragma unroll
        for (int e = 0; e < NEXP; ++e) p[e] *= rs;
        int idx[TOPK]; float val[TOPK]; float vs = 0.f;
#pragma unroll
        for (int k = 0; k < TOPK; ++k) {
            int bi = 0; float bv = -1.f;
#pragma unroll
            for (int e = 0; e < NEXP; ++e)
                if (p[e] > bv) { bv = p[e]; bi = e; }
            idx[k] = bi; val[k] = bv; p[bi] = -2.f; vs += bv;
        }
        float rvs = 1.f / vs;
#pragma unroll
        for (int k = 0; k < TOPK; ++k) {
            topi[t * TOPK + k] = idx[k];
            topv[t * TOPK + k] = val[k] * rvs;
            atomicAdd(&counts[idx[k]], 1);
        }
    }
}

__global__ __launch_bounds__(32) void k_scan(const int* counts, int* offs) {
    if (threadIdx.x == 0) {
        int s = 0;
        for (int e = 0; e < NEXP; ++e) { offs[e] = s; s += counts[e]; }
    }
}

__global__ __launch_bounds__(256) void k_scatter(const int* __restrict__ topi,
                                                 const float* __restrict__ topv,
                                                 const int* __restrict__ offs,
                                                 int* __restrict__ cursor,
                                                 int* __restrict__ btok,
                                                 float* __restrict__ bw) {
    int i = blockIdx.x * 256 + threadIdx.x;
    if (i < T_TOK * TOPK) {
        int e = topi[i];
        int pos = atomicAdd(&cursor[e], 1);
        int j = offs[e] + pos;
        btok[j] = i >> 2;          // token id
        bw[j] = topv[i];           // renormalized gate weight
    }
}

// ---------------- shared expert: gate/up (fused SwiGLU) ----------------
__global__ __launch_bounds__(256) void k_shared_gu(const unsigned short* __restrict__ xb,
                                                   const unsigned short* __restrict__ sgub,
                                                   unsigned short* __restrict__ act) {
    __shared__ __align__(16) unsigned short As[2][BM * LDP];
    __shared__ __align__(16) unsigned short Bg[2][BN * LDP];
    __shared__ __align__(16) unsigned short Bu[2][BN * LDP];
    const int tid = threadIdx.x;
    const int m0 = blockIdx.x * BM, n0 = blockIdx.y * BN;
    const int w = tid >> 5, lane = tid & 31, wmq = w & 3, wnq = w >> 2;
    const int arow = tid >> 1, acol = (tid & 1) * 16;
    const int brow = tid >> 2, bcol = (tid & 3) * 8;

    const unsigned short* axp = xb   + (size_t)(m0 + arow) * HDIM + acol;
    const unsigned short* bgp = sgub + (size_t)(n0 + brow) * HDIM + bcol;
    const unsigned short* bup = sgub + (size_t)(SDIM + n0 + brow) * HDIM + bcol;

    v8f aG[2][2] = {{z8(), z8()}, {z8(), z8()}};
    v8f aU[2][2] = {{z8(), z8()}, {z8(), z8()}};

    ARB ar; BRB brg, bru;
    ld_abf(ar, axp); ld_bbf(brg, bgp); ld_bbf(bru, bup);
    st_abf(As[0], arow, acol, ar);
    st_bbf(Bg[0], brow, bcol, brg);
    st_bbf(Bu[0], brow, bcol, bru);

    const int NS = HDIM / BK;
    for (int s = 0; s < NS; ++s) {
        const int cur = s & 1, nxt = cur ^ 1;
        const int kn = (s + 1) * BK;
        __syncthreads();
        if (s + 1 < NS) {
            ld_abf(ar, axp + kn); ld_bbf(brg, bgp + kn); ld_bbf(bru, bup + kn);
            __builtin_prefetch(bgp + kn + BK, 0, 0);
        }
        mma4(As[cur], Bg[cur], wmq, wnq, lane, aG);
        mma4(As[cur], Bu[cur], wmq, wnq, lane, aU);
        if (s + 1 < NS) {
            st_abf(As[nxt], arow, acol, ar);
            st_bbf(Bg[nxt], brow, bcol, brg);
            st_bbf(Bu[nxt], brow, bcol, bru);
        }
    }
    const int half = lane >> 4, lm = lane & 15;
#pragma unroll
    for (int mt = 0; mt < 2; ++mt)
#pragma unroll
        for (int nt = 0; nt < 2; ++nt)
#pragma unroll
            for (int r = 0; r < 8; ++r) {
                int row = m0 + wmq * 32 + mt * 16 + half * 8 + r;
                int col = n0 + wnq * 32 + nt * 16 + lm;
                act[(size_t)row * SDIM + col] = bf1(siluf(aG[mt][nt][r]) * aU[mt][nt][r]);
            }
}

// ---------------- shared expert: down-proj (writes out) ----------------
__global__ __launch_bounds__(256) void k_shared_down(const unsigned short* __restrict__ act,
                                                     const unsigned short* __restrict__ sdnb,
                                                     float* __restrict__ out) {
    __shared__ __align__(16) unsigned short As[2][BM * LDP];
    __shared__ __align__(16) unsigned short Bs[2][BN * LDP];
    const int tid = threadIdx.x;
    const int m0 = blockIdx.x * BM, n0 = blockIdx.y * BN;
    const int w = tid >> 5, lane = tid & 31, wmq = w & 3, wnq = w >> 2;
    const int arow = tid >> 1, acol = (tid & 1) * 16;
    const int brow = tid >> 2, bcol = (tid & 3) * 8;

    const unsigned short* axp = act  + (size_t)(m0 + arow) * SDIM + acol;
    const unsigned short* bxp = sdnb + (size_t)(n0 + brow) * SDIM + bcol;

    v8f acc[2][2] = {{z8(), z8()}, {z8(), z8()}};
    ARB ar; BRB br;
    ld_abf(ar, axp); ld_bbf(br, bxp);
    st_abf(As[0], arow, acol, ar);
    st_bbf(Bs[0], brow, bcol, br);

    const int NS = SDIM / BK;
    for (int s = 0; s < NS; ++s) {
        const int cur = s & 1, nxt = cur ^ 1;
        const int kn = (s + 1) * BK;
        __syncthreads();
        if (s + 1 < NS) {
            ld_abf(ar, axp + kn); ld_bbf(br, bxp + kn);
            __builtin_prefetch(bxp + kn + BK, 0, 0);
        }
        mma4(As[cur], Bs[cur], wmq, wnq, lane, acc);
        if (s + 1 < NS) {
            st_abf(As[nxt], arow, acol, ar);
            st_bbf(Bs[nxt], brow, bcol, br);
        }
    }
    const int half = lane >> 4, lm = lane & 15;
#pragma unroll
    for (int mt = 0; mt < 2; ++mt)
#pragma unroll
        for (int nt = 0; nt < 2; ++nt)
#pragma unroll
            for (int r = 0; r < 8; ++r) {
                int row = m0 + wmq * 32 + mt * 16 + half * 8 + r;
                int col = n0 + wnq * 32 + nt * 16 + lm;
                out[(size_t)row * HDIM + col] = acc[mt][nt][r];
            }
}

// ---------------- routed experts: h = silu(x_gathered @ w1[e]^T) ----------------
__global__ __launch_bounds__(256) void k_exp_h(const unsigned short* __restrict__ xb,
                                               const unsigned short* __restrict__ w1b,
                                               const int* __restrict__ counts,
                                               const int* __restrict__ offs,
                                               const int* __restrict__ btok,
                                               unsigned short* __restrict__ hact) {
    const int e = blockIdx.x >> 4;
    const int tile = blockIdx.x & 15;
    const int cnt = counts[e];
    if (tile * BM >= cnt) return;
    const int valid = (cnt - tile * BM < BM) ? (cnt - tile * BM) : BM;
    const int slotbase = offs[e] + tile * BM;

    __shared__ int s_tok[BM];
    __shared__ __align__(16) unsigned short As[2][BM * LDP];
    __shared__ __align__(16) unsigned short Bs[2][BN * LDP];
    const int tid = threadIdx.x;
    if (tid < BM) s_tok[tid] = btok[slotbase + ((tid < valid) ? tid : 0)];
    __syncthreads();

    const int n0 = blockIdx.y * BN;
    const unsigned short* wexp = w1b + (size_t)e * IDIM * HDIM;
    const int w = tid >> 5, lane = tid & 31, wmq = w & 3, wnq = w >> 2;
    const int arow = tid >> 1, acol = (tid & 1) * 16;
    const int brow = tid >> 2, bcol = (tid & 3) * 8;

    const unsigned short* axp = xb + (size_t)s_tok[arow] * HDIM + acol;  // clamped gather
    const unsigned short* bxp = wexp + (size_t)(n0 + brow) * HDIM + bcol;

    v8f acc[2][2] = {{z8(), z8()}, {z8(), z8()}};
    ARB ar; BRB br;
    ld_abf(ar, axp); ld_bbf(br, bxp);
    st_abf(As[0], arow, acol, ar);
    st_bbf(Bs[0], brow, bcol, br);

    const int NS = HDIM / BK;
    for (int s = 0; s < NS; ++s) {
        const int cur = s & 1, nxt = cur ^ 1;
        const int kn = (s + 1) * BK;
        __syncthreads();
        if (s + 1 < NS) {
            ld_abf(ar, axp + kn); ld_bbf(br, bxp + kn);
            __builtin_prefetch(bxp + kn + BK, 0, 0);
        }
        mma4(As[cur], Bs[cur], wmq, wnq, lane, acc);
        if (s + 1 < NS) {
            st_abf(As[nxt], arow, acol, ar);
            st_bbf(Bs[nxt], brow, bcol, br);
        }
    }
    const int half = lane >> 4, lm = lane & 15;
#pragma unroll
    for (int mt = 0; mt < 2; ++mt)
#pragma unroll
        for (int nt = 0; nt < 2; ++nt)
#pragma unroll
            for (int r = 0; r < 8; ++r) {
                int row = wmq * 32 + mt * 16 + half * 8 + r;
                if (row < valid) {
                    int col = n0 + wnq * 32 + nt * 16 + lm;
                    hact[(size_t)(slotbase + row) * IDIM + col] = bf1(siluf(acc[mt][nt][r]));
                }
            }
}

// ---------------- routed experts: out += weight * (h @ w2[e]^T) ----------------
__global__ __launch_bounds__(256) void k_exp_out(const unsigned short* __restrict__ hact,
                                                 const unsigned short* __restrict__ w2b,
                                                 const int* __restrict__ counts,
                                                 const int* __restrict__ offs,
                                                 const int* __restrict__ btok,
                                                 const float* __restrict__ bw,
                                                 float* __restrict__ out) {
    const int e = blockIdx.x >> 4;
    const int tile = blockIdx.x & 15;
    const int cnt = counts[e];
    if (tile * BM >= cnt) return;
    const int valid = (cnt - tile * BM < BM) ? (cnt - tile * BM) : BM;
    const int slotbase = offs[e] + tile * BM;

    __shared__ int s_tok[BM];
    __shared__ float s_w[BM];
    __shared__ __align__(16) unsigned short As[2][BM * LDP];
    __shared__ __align__(16) unsigned short Bs[2][BN * LDP];
    const int tid = threadIdx.x;
    if (tid < BM) {
        s_tok[tid] = (tid < valid) ? btok[slotbase + tid] : 0;
        s_w[tid]   = (tid < valid) ? bw[slotbase + tid] : 0.f;
    }
    __syncthreads();

    const int n0 = blockIdx.y * BN;
    const unsigned short* wexp = w2b + (size_t)e * HDIM * IDIM;
    const int w = tid >> 5, lane = tid & 31, wmq = w & 3, wnq = w >> 2;
    const int arow = tid >> 1, acol = (tid & 1) * 16;
    const int brow = tid >> 2, bcol = (tid & 3) * 8;

    const int arowc = (arow < valid) ? arow : 0;   // clamp (rows >= valid unused)
    const unsigned short* axp = hact + (size_t)(slotbase + arowc) * IDIM + acol;
    const unsigned short* bxp = wexp + (size_t)(n0 + brow) * IDIM + bcol;

    v8f acc[2][2] = {{z8(), z8()}, {z8(), z8()}};
    ARB ar; BRB br;
    ld_abf(ar, axp); ld_bbf(br, bxp);
    st_abf(As[0], arow, acol, ar);
    st_bbf(Bs[0], brow, bcol, br);

    const int NS = IDIM / BK;
    for (int s = 0; s < NS; ++s) {
        const int cur = s & 1, nxt = cur ^ 1;
        const int kn = (s + 1) * BK;
        __syncthreads();
        if (s + 1 < NS) {
            ld_abf(ar, axp + kn); ld_bbf(br, bxp + kn);
            __builtin_prefetch(bxp + kn + BK, 0, 0);
        }
        mma4(As[cur], Bs[cur], wmq, wnq, lane, acc);
        if (s + 1 < NS) {
            st_abf(As[nxt], arow, acol, ar);
            st_bbf(Bs[nxt], brow, bcol, br);
        }
    }
    const int half = lane >> 4, lm = lane & 15;
#pragma unroll
    for (int mt = 0; mt < 2; ++mt)
#pragma unroll
        for (int nt = 0; nt < 2; ++nt)
#pragma unroll
            for (int r = 0; r < 8; ++r) {
                int row = wmq * 32 + mt * 16 + half * 8 + r;
                if (row < valid) {
                    int col = n0 + wnq * 32 + nt * 16 + lm;
                    atomicAdd(&out[(size_t)s_tok[row] * HDIM + col],
                              acc[mt][nt][r] * s_w[row]);
                }
            }
}

// ---------------- launcher ----------------
extern "C" void kernel_launch(void* const* d_in, const int* in_sizes, int n_in,
                              void* d_out, int out_size, void* d_ws, size_t ws_size,
                              hipStream_t stream) {
    (void)in_sizes; (void)n_in; (void)out_size; (void)ws_size;
    const float* x   = (const float*)d_in[0];
    const float* gw  = (const float*)d_in[1];
    const float* w1  = (const float*)d_in[2];
    const float* w2  = (const float*)d_in[3];
    const float* sgu = (const float*)d_in[4];
    const float* sdn = (const float*)d_in[5];
    float* out = (float*)d_out;

    // element counts
    const int nX   = T_TOK * HDIM;          // 2M
    const int nSGU = 2 * SDIM * HDIM;       // 2M
    const int nSDN = HDIM * SDIM;           // 1M
    const int nW1  = NEXP * IDIM * HDIM;    // 16M
    const int nW2  = NEXP * HDIM * IDIM;    // 16M

    // workspace layout (~94.3 MiB): bf16 staging copies + act/hact + routing
    char* ws = (char*)d_ws;
    unsigned short* act  = (unsigned short*)ws;                       // [T][SD] bf16
    unsigned short* hact = (unsigned short*)(ws + (size_t)T_TOK * SDIM * 2);
    char* p = ws + (size_t)T_TOK * SDIM * 2 + (size_t)T_TOK * TOPK * IDIM * 2;
    unsigned short* xb   = (unsigned short*)p; p += (size_t)nX * 2;
    unsigned short* sgub = (unsigned short*)p; p += (size_t)nSGU * 2;
    unsigned short* sdnb = (unsigned short*)p; p += (size_t)nSDN * 2;
    unsigned short* w1b  = (unsigned short*)p; p += (size_t)nW1 * 2;
    unsigned short* w2b  = (unsigned short*)p; p += (size_t)nW2 * 2;
    int*   topi   = (int*)p;   p += (size_t)T_TOK * TOPK * 4;
    float* topv   = (float*)p; p += (size_t)T_TOK * TOPK * 4;
    int*   counts = (int*)p;   p += NEXP * 4;
    int*   offs   = (int*)p;   p += NEXP * 4;
    int*   cursor = (int*)p;   p += NEXP * 4;
    int*   btok   = (int*)p;   p += (size_t)T_TOK * TOPK * 4;
    float* bw     = (float*)p;

    // one-shot bf16 preconversion (streaming, ~1.5 us total at HBM rate)
    k_cvt<<<nX   / 2048, 256, 0, stream>>>(x,   xb,   nX);
    k_cvt<<<nSGU / 2048, 256, 0, stream>>>(sgu, sgub, nSGU);
    k_cvt<<<nSDN / 2048, 256, 0, stream>>>(sdn, sdnb, nSDN);
    k_cvt<<<nW1  / 2048, 256, 0, stream>>>(w1,  w1b,  nW1);
    k_cvt<<<nW2  / 2048, 256, 0, stream>>>(w2,  w2b,  nW2);

    // routing
    k_init<<<1, 64, 0, stream>>>(counts, cursor);
    k_router<<<T_TOK / 8, 256, 0, stream>>>(x, gw, topi, topv, counts);
    k_scan<<<1, 32, 0, stream>>>(counts, offs);
    k_scatter<<<(T_TOK * TOPK) / 256, 256, 0, stream>>>(topi, topv, offs, cursor, btok, bw);

    // GEMMs (all bf16-staged, WMMA f32 accumulate)
    k_shared_gu<<<dim3(MTILES, SDIM / BN), 256, 0, stream>>>(xb, sgub, act);
    k_shared_down<<<dim3(MTILES, HDIM / BN), 256, 0, stream>>>(act, sdnb, out);
    k_exp_h<<<dim3(NEXP * MTILES, IDIM / BN), 256, 0, stream>>>(xb, w1b, counts, offs, btok, hact);
    k_exp_out<<<dim3(NEXP * MTILES, HDIM / BN), 256, 0, stream>>>(hact, w2b, counts, offs, btok, bw, out);
}
